// VectorizedLIFLayer_83854941487247
// MI455X (gfx1250) — matrix operand order repaired and checked by reading, not compiled, not run on hardware
//
#include <hip/hip_runtime.h>
#include <stdint.h>

// ---------------- CDNA5 (gfx1250) fused SNN LIF layer ----------------
// Phase A (prep):  W[512,512] f32 -> Wt_hi/Wt_lo[512,512] bf16 (transposed, split)
// Phase B (fused): per (batch, 64-col block):
//   - W block (128 KB split-bf16) async-DMA'd into LDS ONCE per workgroup
//   - spikes chunks (32 KB) double-buffered via GLOBAL_LOAD_ASYNC_TO_LDS_B128,
//     DMA overlapping the WMMAs of the current chunk (ASYNCcnt)
//   - 16x64 GEMM chunks via V_WMMA_F32_16X16X32_BF16 (split-bf16 = 3 WMMA/K-step),
//     ALL operands fed from LDS (ds_load_b128 only in the hot loop)
//   - in-register LIF scan across all 1024 timesteps

typedef __attribute__((ext_vector_type(16))) __bf16        v16bf;
typedef __attribute__((ext_vector_type(8)))  float         v8f;
typedef __attribute__((ext_vector_type(4)))  unsigned int  u32x4;

#define BATCH  32
#define TIME   1024
#define IN_DIM 512
#define HID    512
#define NB     64    // hidden columns per workgroup (4 waves x 16)
#define MC     16    // timesteps (GEMM rows) per chunk
#define KPAD   520   // padded bf16 row stride: 1040 B (16B aligned, bank-skewed)

union FragBF { v16bf v; u32x4 q[2]; };

__device__ __forceinline__ void split_bf16(float w, __bf16& hi, __bf16& lo) {
    hi = (__bf16)w;                 // RNE truncation to bf16
    lo = (__bf16)(w - (float)hi);   // residual (captures ~8 more mantissa bits)
}

__device__ __forceinline__ void wait_async0() {
    asm volatile("s_wait_asynccnt 0" ::: "memory");
}

// One 16-byte async DMA lane transfer: global -> LDS (tracked by ASYNCcnt).
__device__ __forceinline__ void async_b128(const void* g, const void* lds_generic) {
    unsigned lds = (unsigned)(uintptr_t)
        (__attribute__((address_space(3))) const char*)lds_generic;
    asm volatile("global_load_async_to_lds_b128 %0, %1, off"
                 :: "v"(lds), "v"(g) : "memory");
}

// Issue async DMA of spikes[b, t0:t0+16, :] (16x512 f32 = 32 KB) into LDS staging.
__device__ __forceinline__ void async_stage(const float* __restrict__ spikes,
                                            int b, int t0, int tid,
                                            float (*sRaw)[IN_DIM]) {
#pragma unroll
    for (int j = 0; j < (MC * IN_DIM / 4) / 128; ++j) {   // 16 iterations
        int idx4 = tid + 128 * j;        // 0..2047 float4s
        int t    = idx4 >> 7;            // 128 float4 per row
        int kq   = idx4 & 127;
        async_b128(spikes + (size_t)(b * TIME + t0 + t) * IN_DIM + kq * 4,
                   &sRaw[t][kq * 4]);
    }
}

// ---- Phase A: transpose + hi/lo split of W into workspace ----
__global__ void snn_prep_w(const float* __restrict__ W,
                           __bf16* __restrict__ Wt_hi,
                           __bf16* __restrict__ Wt_lo) {
    int i = blockIdx.x * blockDim.x + threadIdx.x;  // over 512*512
    int k = i / HID;
    int h = i % HID;
    float w = W[i];
    __bf16 hi, lo;
    split_bf16(w, hi, lo);
    Wt_hi[(size_t)h * IN_DIM + k] = hi;             // column-major wrt W => B-frag contiguous
    Wt_lo[(size_t)h * IN_DIM + k] = lo;
}

// ---- Phase B: fused GEMM (WMMA bf16-split) + LIF scan ----
__global__ __launch_bounds__(128)
void snn_lif_fused(const float*  __restrict__ spikes,  // [B, T, IN] f32
                   const __bf16* __restrict__ Wt_hi,   // [H, IN] bf16
                   const __bf16* __restrict__ Wt_lo,   // [H, IN] bf16
                   const float*  __restrict__ bias,    // [H] f32
                   float*        __restrict__ out) {   // [B, T, H] f32
    __shared__ float  sRaw[MC][IN_DIM];                // async DMA staging (32 KB)
    __shared__ __bf16 sAhi[MC][KPAD];                  // split-bf16 A (2 x 16.3 KB)
    __shared__ __bf16 sAlo[MC][KPAD];
    __shared__ __bf16 sBhi[NB][KPAD];                  // split-bf16 W block (2 x 65 KB)
    __shared__ __bf16 sBlo[NB][KPAD];
    __shared__ float  sI[MC][NB + 4];                  // +4 dwords: skew banks

    const int b     = blockIdx.y;
    const int h0    = blockIdx.x * NB;
    const int tid   = threadIdx.x;
    const int lane  = tid & 31;
    const int wave  = tid >> 5;       // 0..3, each owns one 16-wide n-tile
    const int nloc  = lane & 15;
    const int khalf = lane >> 4;      // 0 | 1 (wave32 half)

    // LIF state lives in registers of the first 64 threads for the whole T loop.
    float v_mem = 0.0f, i_syn = 0.0f, bia = 0.0f;
    if (tid < NB) bia = bias[h0 + tid];

    const float a_m = 0.95122942450071400910f;  // exp(-DT/TAU_MEM) = exp(-0.05)
    const float a_s = 0.81873075307798185867f;  // exp(-DT/TAU_SYN) = exp(-0.2)

    // --- Prologue: async-DMA the whole W block (hi+lo) into LDS, reused by
    //     all 64 chunks; also kick off spikes chunk 0. One wait covers all.
#pragma unroll
    for (int j = 0; j < (NB * IN_DIM / 8) / 128; ++j) {   // 32 iters x 2 arrays
        int idx = tid + 128 * j;         // 0..4095 16B-chunks (8 bf16 each)
        int h   = idx >> 6;              // 64 chunks per row
        int kc  = (idx & 63) * 8;
        async_b128(Wt_hi + (size_t)(h0 + h) * IN_DIM + kc, &sBhi[h][kc]);
        async_b128(Wt_lo + (size_t)(h0 + h) * IN_DIM + kc, &sBlo[h][kc]);
    }
    async_stage(spikes, b, 0, tid, sRaw);

    for (int t0 = 0; t0 < TIME; t0 += MC) {
        // --- staging for this chunk (and W on iter 0) complete ---
        wait_async0();
        __syncthreads();

        // --- convert staged f32 -> split-bf16 fragment layout ---
        #pragma unroll
        for (int j = 0; j < (MC * IN_DIM / 4) / 128; ++j) {  // 16 iterations
            int idx4 = tid + 128 * j;
            int t    = idx4 >> 7;
            int kq   = idx4 & 127;
            const float4 sv = *(const float4*)&sRaw[t][kq * 4];
            int k = kq * 4;
            __bf16 hi, lo;
            split_bf16(sv.x, hi, lo); sAhi[t][k + 0] = hi; sAlo[t][k + 0] = lo;
            split_bf16(sv.y, hi, lo); sAhi[t][k + 1] = hi; sAlo[t][k + 1] = lo;
            split_bf16(sv.z, hi, lo); sAhi[t][k + 2] = hi; sAlo[t][k + 2] = lo;
            split_bf16(sv.w, hi, lo); sAhi[t][k + 3] = hi; sAlo[t][k + 3] = lo;
        }
        __syncthreads();   // sAhi/sAlo published; sRaw free for next chunk's DMA

        // --- overlap: issue DMA for chunk i+1 while this chunk's WMMAs run ---
        if (t0 + MC < TIME) async_stage(spikes, b, t0 + MC, tid, sRaw);

        // --- GEMM: 16x16 tile per wave over K=512, split-bf16 (3 WMMA per K-step),
        //     all operands from LDS (pure ds_load_b128 + v_wmma hot loop) ---
        v8f C = {};
        const int nw = wave * 16 + nloc;
        #pragma unroll 4
        for (int ks = 0; ks < IN_DIM; ks += 32) {
            FragBF ahi, alo, bhi, blo;
            // A fragment (ISA 16-bit A 16x32 layout): lane half0 -> K{0..7,16..23},
            // half1 -> K{8..15,24..31}; row M = lane%16. Two ds_load_b128 each.
            const int ka = ks + khalf * 8;
            ahi.q[0] = *(const u32x4*)&sAhi[nloc][ka];
            ahi.q[1] = *(const u32x4*)&sAhi[nloc][ka + 16];
            alo.q[0] = *(const u32x4*)&sAlo[nloc][ka];
            alo.q[1] = *(const u32x4*)&sAlo[nloc][ka + 16];
            // B fragment (32x16): lane half0 -> K{0..7,16..23}, half1 -> K{8..15,24..31}, col N=lane%16.
            const int kb = ks + khalf * 8;
            bhi.q[0] = *(const u32x4*)&sBhi[nw][kb];
            bhi.q[1] = *(const u32x4*)&sBhi[nw][kb + 16];
            blo.q[0] = *(const u32x4*)&sBlo[nw][kb];
            blo.q[1] = *(const u32x4*)&sBlo[nw][kb + 16];

            C = __builtin_amdgcn_wmma_f32_16x16x32_bf16(false, ahi.v, false, bhi.v,
                                                        (short)0, C, false, false);
            C = __builtin_amdgcn_wmma_f32_16x16x32_bf16(false, ahi.v, false, blo.v,
                                                        (short)0, C, false, false);
            C = __builtin_amdgcn_wmma_f32_16x16x32_bf16(false, alo.v, false, bhi.v,
                                                        (short)0, C, false, false);
        }

        // --- spill D tile to LDS (f32 C/D layout: VGPR r -> M = r + 8*khalf, N = lane%16)
        {
            const int m0 = khalf * 8;
            #pragma unroll
            for (int r = 0; r < 8; ++r) sI[m0 + r][nw] = C[r];
        }
        __syncthreads();   // sI published (next-chunk DMA keeps flowing underneath)

        // --- LIF scan over the 16 new timesteps (state in registers) ---
        if (tid < NB) {
            float* orow = out + (size_t)(b * TIME + t0) * HID + h0 + tid;
            #pragma unroll
            for (int t = 0; t < MC; ++t) {
                float I = sI[t][tid] + bia;
                i_syn = a_s * i_syn + I;
                v_mem = a_m * v_mem + i_syn;
                float s = (v_mem > 1.0f) ? 1.0f : 0.0f;
                v_mem -= s;                      // soft reset
                orow[(size_t)t * HID] = s;
            }
        }
        __syncthreads();                         // protect sI/sAhi reuse next chunk
    }
}

extern "C" void kernel_launch(void* const* d_in, const int* in_sizes, int n_in,
                              void* d_out, int out_size, void* d_ws, size_t ws_size,
                              hipStream_t stream) {
    const float* spikes = (const float*)d_in[0];   // [32,1024,512] f32
    const float* W      = (const float*)d_in[1];   // [512,512] f32
    const float* bias   = (const float*)d_in[2];   // [512] f32
    float*       out    = (float*)d_out;           // [32,1024,512] f32

    // workspace: 512 KB hi + 512 KB lo split-bf16 transposed W
    __bf16* Wt_hi = (__bf16*)d_ws;
    __bf16* Wt_lo = Wt_hi + (size_t)IN_DIM * HID;

    snn_prep_w<<<dim3((IN_DIM * HID) / 256), dim3(256), 0, stream>>>(W, Wt_hi, Wt_lo);
    snn_lif_fused<<<dim3(HID / NB, BATCH), dim3(128), 0, stream>>>(
        spikes, Wt_hi, Wt_lo, bias, out);
}
